// Attn_6897717477791
// MI455X (gfx1250) — matrix operand order repaired and checked by reading, not compiled, not run on hardware
//
#include <hip/hip_runtime.h>
#include <hip/hip_bf16.h>

// Problem constants (match reference)
constexpr int BATCH = 16;
constexpr int LQN   = 2048;
constexpr int LKN   = 2048;
constexpr int DH    = 128;   // head dim
constexpr int QB    = 32;    // queries per workgroup
constexpr int KT    = 64;    // keys per k-tile
constexpr int NTHR  = 256;   // 8 wave32

typedef __attribute__((ext_vector_type(16))) __bf16 v16bf;
typedef __attribute__((ext_vector_type(8)))  float  v8f;

union FragA { v16bf v; uint4 u[2]; };
union FragC { v8f   v; float f[8]; };

__device__ __forceinline__ unsigned pk_bf16(float a, float b) {
    union { __bf16 h[2]; unsigned u; } t;
    t.h[0] = (__bf16)a; t.h[1] = (__bf16)b;
    return t.u;
}

__global__ __launch_bounds__(NTHR)
void attn_fwd_wmma(const float* __restrict__ Q,
                   const float* __restrict__ K,
                   const float* __restrict__ V,
                   float* __restrict__ Rout,
                   float* __restrict__ Aout)
{
    // LDS staging (~45 KB of the 320 KB WGP pool)
    __shared__ __bf16 Qs[QB * DH];        // Q tile, pre-scaled, row-major   (8 KB)
    __shared__ __bf16 KTs[DH * KT];       // K tile row-major: [key][dim]   (16 KB)
    __shared__ __bf16 Vs[KT * DH];        // V tile transposed: [dim][key]    (16 KB)
    __shared__ __bf16 Pb[QB * KT];        // P exchange buffer               (4 KB)
    __shared__ float  mpart[4][QB], lpart[4][QB];
    __shared__ float  mrow[QB], linv[QB];

    const int tid  = threadIdx.x;
    const int lane = tid & 31;
    const int w    = tid >> 5;   // wave id 0..7
    const int rb   = w >> 2;     // S-phase row block (q 0-15 / 16-31)
    const int cb   = w & 3;      // S-phase col block (16 keys each)
    const int b    = blockIdx.x >> 6;          // 64 query-blocks per batch
    const int q0   = (blockIdx.x & 63) * QB;

    const float scale = 0.08838834764831845f;  // 1/sqrt(128)

    const float* Qg = Q + ((size_t)b * LQN + q0) * DH;
    const float* Kg = K + (size_t)b * LKN * DH;
    const float* Vg = V + (size_t)b * LKN * DH;

    // ---- stage Q (pre-scaled) as bf16 ----
    #pragma unroll
    for (int i = 0; i < (QB * DH / 4) / NTHR; ++i) {
        int flat = tid + i * NTHR;
        int row = flat >> 5, d4 = flat & 31;
        float4 qv = reinterpret_cast<const float4*>(Qg)[(size_t)row * (DH / 4) + d4];
        unsigned* dst = reinterpret_cast<unsigned*>(&Qs[row * DH + d4 * 4]);
        dst[0] = pk_bf16(qv.x * scale, qv.y * scale);
        dst[1] = pk_bf16(qv.z * scale, qv.w * scale);
    }
    __syncthreads();

    // ---- per-wave A-fragments of Q (16x32 bf16 layout, ISA 7.12.2) ----
    FragA qa[4];
    {
        int row = rb * 16 + (lane & 15);
        int kb0 = (lane & 16) ? 8 : 0;
        #pragma unroll
        for (int kc = 0; kc < 4; ++kc) {
            const __bf16* p = &Qs[row * DH + kc * 32 + kb0];
            qa[kc].u[0] = *reinterpret_cast<const uint4*>(p);
            qa[kc].u[1] = *reinterpret_cast<const uint4*>(p + 16);
        }
    }

    // =========== Pass A: per-lane online row max / sum(exp) ===========
    // Each lane sees exactly one element per C-VGPR per k-tile (its private
    // column stream of the row), so stats accumulate lane-locally with no
    // cross-lane traffic; one shuffle-tree merge happens after the loop.
    float mst[8], lst[8];
    #pragma unroll
    for (int v = 0; v < 8; ++v) { mst[v] = -__builtin_inff(); lst[v] = 0.f; }

    for (int k0 = 0; k0 < LKN; k0 += KT) {
        __syncthreads();
        // stage K^T tile as bf16
        #pragma unroll
        for (int i = 0; i < (KT * DH / 4) / NTHR; ++i) {
            int flat = tid + i * NTHR;
            int key = flat >> 5, d4 = flat & 31;
            float4 kv = reinterpret_cast<const float4*>(Kg)[(size_t)(k0 + key) * (DH / 4) + d4];
            KTs[key * DH + d4 * 4 + 0] = (__bf16)kv.x;
            KTs[key * DH + d4 * 4 + 1] = (__bf16)kv.y;
            KTs[key * DH + d4 * 4 + 2] = (__bf16)kv.z;
            KTs[key * DH + d4 * 4 + 3] = (__bf16)kv.w;
        }
        if (k0 + KT < LKN) {  // pull next K tile toward L2 (global_prefetch_b8)
            const char* nxt = reinterpret_cast<const char*>(Kg + (size_t)(k0 + KT) * DH);
            __builtin_prefetch(nxt + (size_t)tid * 128, 0, 1);
        }
        __syncthreads();

        // S tile = (scale*Q) @ K^T  for this wave's (rb, cb)
        FragC c;
        #pragma unroll
        for (int v = 0; v < 8; ++v) c.f[v] = 0.f;
        #pragma unroll
        for (int kc = 0; kc < 4; ++kc) {
            FragA kb;
            const __bf16* p = &KTs[(cb * 16 + (lane & 15)) * DH + kc * 32 + ((lane & 16) ? 8 : 0)];
            kb.u[0] = *reinterpret_cast<const uint4*>(p);
            kb.u[1] = *reinterpret_cast<const uint4*>(p + 16);
            c.v = __builtin_amdgcn_wmma_f32_16x16x32_bf16(
                      false, qa[kc].v, false, kb.v, (short)0, c.v, false, false);
        }

        // lane-local online update: 2 TRANS exp + max + fma per element
        #pragma unroll
        for (int v = 0; v < 8; ++v) {
            float s  = c.f[v];
            float mo = mst[v];
            float mn = fmaxf(mo, s);
            lst[v] = lst[v] * __expf(mo - mn) + __expf(s - mn);
            mst[v] = mn;
        }
    }

    // one-time cross-lane merge of (m,l) pairs within each 16-lane row group
    #pragma unroll
    for (int v = 0; v < 8; ++v) {
        float m = mst[v], l = lst[v];
        #pragma unroll
        for (int off = 8; off >= 1; off >>= 1) {
            float mo = __shfl_xor(m, off, 16);
            float lo = __shfl_xor(l, off, 16);
            float mn = fmaxf(m, mo);
            l = l * __expf(m - mn) + lo * __expf(mo - mn);
            m = mn;
        }
        int row = rb * 16 + ((lane & 16) ? 8 : 0) + v;
        if ((lane & 15) == 0) { mpart[cb][row] = m; lpart[cb][row] = l; }
    }
    __syncthreads();

    // combine the 4 col-block waves per row
    if (tid < QB) {
        float m = fmaxf(fmaxf(mpart[0][tid], mpart[1][tid]),
                        fmaxf(mpart[2][tid], mpart[3][tid]));
        float l = 0.f;
        #pragma unroll
        for (int j = 0; j < 4; ++j) l += lpart[j][tid] * __expf(mpart[j][tid] - m);
        mrow[tid] = m;
        linv[tid] = 1.f / l;
    }
    __syncthreads();

    // per-lane row constants for S-phase rows
    float mr[8], lr[8];
    #pragma unroll
    for (int v = 0; v < 8; ++v) {
        int row = rb * 16 + ((lane & 16) ? 8 : 0) + v;
        mr[v] = mrow[row];
        lr[v] = linv[row];
    }

    // =========== Pass B: recompute S, emit P once, O += P @ V ===========
    FragC o0, o1;  // wave owns dim-block db = w, for both query row-blocks
    #pragma unroll
    for (int v = 0; v < 8; ++v) { o0.f[v] = 0.f; o1.f[v] = 0.f; }

    for (int k0 = 0; k0 < LKN; k0 += KT) {
        __syncthreads();
        // stage K^T and V tiles as bf16
        #pragma unroll
        for (int i = 0; i < (KT * DH / 4) / NTHR; ++i) {
            int flat = tid + i * NTHR;
            int key = flat >> 5, d4 = flat & 31;
            float4 kv = reinterpret_cast<const float4*>(Kg)[(size_t)(k0 + key) * (DH / 4) + d4];
            KTs[key * DH + d4 * 4 + 0] = (__bf16)kv.x;
            KTs[key * DH + d4 * 4 + 1] = (__bf16)kv.y;
            KTs[key * DH + d4 * 4 + 2] = (__bf16)kv.z;
            KTs[key * DH + d4 * 4 + 3] = (__bf16)kv.w;
            float4 vv = reinterpret_cast<const float4*>(Vg)[(size_t)(k0 + key) * (DH / 4) + d4];
            Vs[(d4 * 4 + 0) * KT + key] = (__bf16)vv.x;
            Vs[(d4 * 4 + 1) * KT + key] = (__bf16)vv.y;
            Vs[(d4 * 4 + 2) * KT + key] = (__bf16)vv.z;
            Vs[(d4 * 4 + 3) * KT + key] = (__bf16)vv.w;
        }
        if (k0 + KT < LKN) {
            const char* nxt = reinterpret_cast<const char*>(Vg + (size_t)(k0 + KT) * DH);
            __builtin_prefetch(nxt + (size_t)tid * 128, 0, 1);
        }
        __syncthreads();

        // S tile recompute
        FragC c;
        #pragma unroll
        for (int v = 0; v < 8; ++v) c.f[v] = 0.f;
        #pragma unroll
        for (int kc = 0; kc < 4; ++kc) {
            FragA kb;
            const __bf16* p = &KTs[(cb * 16 + (lane & 15)) * DH + kc * 32 + ((lane & 16) ? 8 : 0)];
            kb.u[0] = *reinterpret_cast<const uint4*>(p);
            kb.u[1] = *reinterpret_cast<const uint4*>(p + 16);
            c.v = __builtin_amdgcn_wmma_f32_16x16x32_bf16(
                      false, qa[kc].v, false, kb.v, (short)0, c.v, false, false);
        }

        // P = exp(S - m) / l : write attn output once, and to LDS for P@V
        #pragma unroll
        for (int v = 0; v < 8; ++v) {
            float p = __expf(c.f[v] - mr[v]) * lr[v];
            int row = rb * 16 + ((lane & 16) ? 8 : 0) + v;
            int kl  = cb * 16 + (lane & 15);
            Aout[((size_t)b * LQN + q0 + row) * LKN + (k0 + kl)] = p;
            Pb[row * KT + kl] = (__bf16)p;
        }
        __syncthreads();

        // O(rb2, db=w) += P(rb2, keychunk) @ V(keychunk, db)
        #pragma unroll
        for (int kc = 0; kc < 2; ++kc) {
            FragA vbf;
            const __bf16* vp = &Vs[(w * 16 + (lane & 15)) * KT + kc * 32 + ((lane & 16) ? 8 : 0)];
            vbf.u[0] = *reinterpret_cast<const uint4*>(vp);
            vbf.u[1] = *reinterpret_cast<const uint4*>(vp + 16);

            FragA pa0, pa1;
            int prow0 = (lane & 15);
            int prow1 = 16 + (lane & 15);
            int kb0 = kc * 32 + ((lane & 16) ? 8 : 0);
            const __bf16* pp0 = &Pb[prow0 * KT + kb0];
            const __bf16* pp1 = &Pb[prow1 * KT + kb0];
            pa0.u[0] = *reinterpret_cast<const uint4*>(pp0);
            pa0.u[1] = *reinterpret_cast<const uint4*>(pp0 + 16);
            pa1.u[0] = *reinterpret_cast<const uint4*>(pp1);
            pa1.u[1] = *reinterpret_cast<const uint4*>(pp1 + 16);

            o0.v = __builtin_amdgcn_wmma_f32_16x16x32_bf16(
                       false, pa0.v, false, vbf.v, (short)0, o0.v, false, false);
            o1.v = __builtin_amdgcn_wmma_f32_16x16x32_bf16(
                       false, pa1.v, false, vbf.v, (short)0, o1.v, false, false);
        }
    }

    // ---- write R (already normalized since P was normalized) ----
    #pragma unroll
    for (int v = 0; v < 8; ++v) {
        int dcol = w * 16 + (lane & 15);
        int row0 = ((lane & 16) ? 8 : 0) + v;
        int row1 = 16 + row0;
        Rout[((size_t)b * LQN + q0 + row0) * DH + dcol] = o0.f[v];
        Rout[((size_t)b * LQN + q0 + row1) * DH + dcol] = o1.f[v];
    }
}

extern "C" void kernel_launch(void* const* d_in, const int* in_sizes, int n_in,
                              void* d_out, int out_size, void* d_ws, size_t ws_size,
                              hipStream_t stream) {
    const float* Q = (const float*)d_in[0];
    const float* K = (const float*)d_in[1];
    const float* V = (const float*)d_in[2];
    float* Rout = (float*)d_out;                           // [B, LQ, D]
    float* Aout = Rout + (size_t)BATCH * LQN * DH;         // [B, LQ, LK]
    dim3 grid(BATCH * (LQN / QB));                         // 1024 workgroups
    attn_fwd_wmma<<<grid, NTHR, 0, stream>>>(Q, K, V, Rout, Aout);
}
